// BandwidthConstrainedComm_84035330113932
// MI455X (gfx1250) — matrix operand order repaired and checked by reading, not compiled, not run on hardware
//
#include <hip/hip_runtime.h>
#include <math.h>

typedef __attribute__((ext_vector_type(16))) _Float16 v16h;
typedef __attribute__((ext_vector_type(8)))  _Float16 v8h;
typedef __attribute__((ext_vector_type(8)))  float    v8f;

// Dimensions
#define BDIM 128
#define NAG  256
#define DDIM 256
#define MSGD 64
#define CDD  32
#define KTOP 8
#define H1D  128
#define H2D  256
#define MROWS (BDIM * NAG)   // 32768

#define FLAG_RELU  1
#define FLAG_F16O  2

// ---------------------------------------------------------------------------
// Elementwise f32 -> f16 (for obs). 4 elements/thread.
// ---------------------------------------------------------------------------
__global__ __launch_bounds__(256)
void cvt_f32_to_f16(const float* __restrict__ in, _Float16* __restrict__ out)
{
    const size_t i = ((size_t)blockIdx.x * 256 + threadIdx.x) * 4;
#pragma unroll
    for (int t = 0; t < 4; ++t) out[i + t] = (_Float16)in[i + t];
}

// ---------------------------------------------------------------------------
// Pack a row-major f32 weight W[K,N] into WMMA B-fragment-major f16:
// frag index f = (kb/32)*(N/16) + ntile ; per lane 16 contiguous halfs where
// b[h] = W[kb + 16*(lane/16) + h][ntile*16 + lane%16].
// One wave per fragment. grid = (K/32)*(N/16) blocks of 32 threads.
// ---------------------------------------------------------------------------
__global__ __launch_bounds__(32)
void pack_b_frag(const float* __restrict__ W, _Float16* __restrict__ out,
                 int N, int ntiles)
{
    const int frag = blockIdx.x;
    const int lane = threadIdx.x;
    const int ks   = frag / ntiles;
    const int nt   = frag % ntiles;
    const int l16  = lane & 15;
    const int half = lane >> 4;

    const float* wp = W + (size_t)(ks * 32 + half * 16) * N + nt * 16 + l16;
    _Float16*    op = out + ((size_t)frag * 32 + lane) * 16;
#pragma unroll
    for (int h = 0; h < 16; ++h) op[h] = (_Float16)wp[(size_t)h * N];
}

// ---------------------------------------------------------------------------
// K-loop body: straightline address advance, one A fragment amortized over
// NT WMMA n-tiles. arow points at the A source row; k indices [k0,k1) are
// logical, koff rebases into the source.
// ---------------------------------------------------------------------------
template<int NT>
__device__ __forceinline__
void gemm_k_loop(const _Float16* __restrict__ arow, int k0, int k1, int koff,
                 const v16h* __restrict__ bbase, int nfrags, int half,
                 v8f (&c)[NT])
{
    for (int kb = k0; kb < k1; kb += 32) {
        const _Float16* ap = arow + (kb - koff) + half * 16;
        const v8h lo = *(const v8h*)(ap);
        const v8h hi = *(const v8h*)(ap + 8);
        v16h a;
#pragma unroll
        for (int i = 0; i < 8; ++i) { a[i] = lo[i]; a[i + 8] = hi[i]; }

        const v16h* bp = bbase + (size_t)(kb >> 5) * nfrags * 32;
#pragma unroll
        for (int j = 0; j < NT; ++j) {
            const v16h b = bp[j * 32];
            c[j] = __builtin_amdgcn_wmma_f32_16x16x32_f16(
                       false, a, false, b, (short)0, c[j], false, false);
        }
    }
}

// ---------------------------------------------------------------------------
// WMMA GEMM: out[M,N] = act( A[M,K] @ W[K,N] + bias )
// A: f16 row-major, split at k=ksplit between A0/A1 (fused concat).
// B: pre-packed fragment-major f16 (see pack_b_frag).
// Each wave computes a 16 x (16*NT) output block. 8 waves/block.
// grid = (M/128, N/(16*NT)).
// ---------------------------------------------------------------------------
template<int NT>
__global__ __launch_bounds__(256)
void wmma_gemm(const _Float16* __restrict__ A0, int lda0,
               const _Float16* __restrict__ A1, int lda1, int ksplit,
               const _Float16* __restrict__ Bpack,
               const float* __restrict__ bias,
               void* __restrict__ out,
               int K, int N, int flags)
{
    const int lane = threadIdx.x & 31;
    const int wave = threadIdx.x >> 5;
    const int tm   = (blockIdx.x * 8 + wave) << 4;
    const int tn   = blockIdx.y * (16 * NT);
    const int l16  = lane & 15;
    const int half = lane >> 4;
    const int mA   = tm + l16;
    const int nfrags = N >> 4;

    const v16h* bbase = (const v16h*)Bpack + ((size_t)(tn >> 4) * 32 + lane);

    v8f c[NT];
#pragma unroll
    for (int j = 0; j < NT; ++j) c[j] = (v8f){};

    const int kmid = (ksplit < K) ? ksplit : K;
    gemm_k_loop<NT>(A0 + (size_t)mA * lda0, 0, kmid, 0, bbase, nfrags, half, c);
    if (kmid < K)
        gemm_k_loop<NT>(A1 + (size_t)mA * lda1, kmid, K, kmid, bbase, nfrags, half, c);

#pragma unroll
    for (int j = 0; j < NT; ++j) {
        const int n = tn + j * 16 + l16;
        const float bv = bias ? bias[n] : 0.0f;
        if (flags & FLAG_F16O) {
            _Float16* o = (_Float16*)out;
#pragma unroll
            for (int i = 0; i < 8; ++i) {
                float v = c[j][i] + bv;
                if (flags & FLAG_RELU) v = fmaxf(v, 0.0f);
                o[(size_t)(tm + i + half * 8) * N + n] = (_Float16)v;
            }
        } else {
            float* o = (float*)out;
#pragma unroll
            for (int i = 0; i < 8; ++i) {
                float v = c[j][i] + bv;
                if (flags & FLAG_RELU) v = fmaxf(v, 0.0f);
                o[(size_t)(tm + i + half * 8) * N + n] = v;
            }
        }
    }
}

// ---------------------------------------------------------------------------
// Pairwise bilinear scores: scores[b,i,j] = tmp16[b,i,:] . obs16[b,j,:] + bbil
// B-matrix is obs^T, so tile-j fragment b[h] = obs16[b, tn+16j+n, kb+16*half+h]
// (contiguous 32B per-lane loads). Each wave: 16x64 output block.
// grid = (NAG/16/8, NAG/64, BDIM).
// ---------------------------------------------------------------------------
__global__ __launch_bounds__(256)
void wmma_scores(const _Float16* __restrict__ tmp16,
                 const _Float16* __restrict__ obs16,
                 const float* __restrict__ bbil,
                 float* __restrict__ scores)
{
    const int lane = threadIdx.x & 31;
    const int wave = threadIdx.x >> 5;
    const int bidx = blockIdx.z;
    const int tm   = (blockIdx.x * 8 + wave) << 4;
    const int tn   = blockIdx.y << 6;                 // 64 cols per wave
    const int l16  = lane & 15;
    const int half = lane >> 4;

    const _Float16* Ar = tmp16 + ((size_t)bidx * NAG + tm + l16) * DDIM;
    const _Float16* Br = obs16 + ((size_t)bidx * NAG + tn + l16) * DDIM;

    v8f c[4];
#pragma unroll
    for (int j = 0; j < 4; ++j) c[j] = (v8f){};

    for (int kb = 0; kb < DDIM; kb += 32) {
        const v8h lo = *(const v8h*)(Ar + kb + half * 16);
        const v8h hi = *(const v8h*)(Ar + kb + half * 16 + 8);
        v16h a;
#pragma unroll
        for (int i = 0; i < 8; ++i) { a[i] = lo[i]; a[i + 8] = hi[i]; }

#pragma unroll
        for (int j = 0; j < 4; ++j) {
            const v16h b = *(const v16h*)(Br + (size_t)j * 16 * DDIM + kb + half * 16);
            c[j] = __builtin_amdgcn_wmma_f32_16x16x32_f16(
                       false, a, false, b, (short)0, c[j], false, false);
        }
    }

    const float bb = bbil[0];
#pragma unroll
    for (int j = 0; j < 4; ++j)
#pragma unroll
        for (int i = 0; i < 8; ++i)
            scores[((size_t)bidx * NAG + tm + i + half * 8) * NAG + tn + j * 16 + l16] =
                c[j][i] + bb;
}

// ---------------------------------------------------------------------------
// Fused top-8 + softmax + gated aggregation. One wave per receiver row.
// grid = MROWS/8 blocks of 256 threads.
// ---------------------------------------------------------------------------
__global__ __launch_bounds__(256)
void topk_softmax_agg(const float* __restrict__ scores,
                      const _Float16* __restrict__ msgs16,
                      _Float16* __restrict__ agg16)
{
    const int lane = threadIdx.x & 31;
    const int wave = threadIdx.x >> 5;
    const int row  = blockIdx.x * 8 + wave;       // b*NAG + i
    const float* s = scores + (size_t)row * NAG;

    float v[8];
#pragma unroll
    for (int t = 0; t < 8; ++t) v[t] = s[lane * 8 + t];

    unsigned sel = 0;
    float topv[KTOP];
    int   topi[KTOP];

    for (int k = 0; k < KTOP; ++k) {
        float best = -__builtin_inff();
        int   bi   = 0x7FFFFFFF;
#pragma unroll
        for (int t = 0; t < 8; ++t) {
            if (!(sel & (1u << t))) {
                const int j = lane * 8 + t;
                if (v[t] > best || (v[t] == best && j < bi)) { best = v[t]; bi = j; }
            }
        }
        for (int off = 16; off; off >>= 1) {   // wave32 argmax, low-index ties
            const float ov = __shfl_xor(best, off, 32);
            const int   oi = __shfl_xor(bi,   off, 32);
            if (ov > best || (ov == best && oi < bi)) { best = ov; bi = oi; }
        }
        topv[k] = best;
        topi[k] = bi;
        if ((bi >> 3) == lane) sel |= 1u << (bi & 7);
    }

    const float mx = topv[0];                   // first pick is global max
    float e[KTOP], sum = 0.0f;
#pragma unroll
    for (int k = 0; k < KTOP; ++k) { e[k] = expf(topv[k] - mx); sum += e[k]; }
    const float inv = 1.0f / sum;

    const int bb = row >> 8;
    float a0 = 0.0f, a1 = 0.0f;
#pragma unroll
    for (int k = 0; k < KTOP; ++k) {
        const float g = e[k] * inv;
        const _Float16* mp = msgs16 + ((size_t)(bb << 8) + topi[k]) * MSGD;
        a0 += g * (float)mp[lane];
        a1 += g * (float)mp[lane + 32];
    }
    agg16[(size_t)row * MSGD + lane]      = (_Float16)a0;
    agg16[(size_t)row * MSGD + 32 + lane] = (_Float16)a1;
}

// ---------------------------------------------------------------------------
// Host-side orchestration
// ---------------------------------------------------------------------------
extern "C" void kernel_launch(void* const* d_in, const int* in_sizes, int n_in,
                              void* d_out, int out_size, void* d_ws, size_t ws_size,
                              hipStream_t stream)
{
    const float* obs  = (const float*)d_in[0];
    const float* W1   = (const float*)d_in[1];
    const float* b1   = (const float*)d_in[2];
    const float* W2   = (const float*)d_in[3];
    const float* b2   = (const float*)d_in[4];
    const float* Wc   = (const float*)d_in[5];
    const float* bc   = (const float*)d_in[6];
    const float* Wd   = (const float*)d_in[7];
    const float* bd   = (const float*)d_in[8];
    const float* Wbil = (const float*)d_in[9];
    const float* bbil = (const float*)d_in[10];
    const float* Wr1  = (const float*)d_in[11];
    const float* br1  = (const float*)d_in[12];
    const float* Wr2  = (const float*)d_in[13];
    const float* br2  = (const float*)d_in[14];
    float* out = (float*)d_out;

    // ---- workspace carving (64B aligned regions) ----
    char* w = (char*)d_ws;
    auto carve = [&](size_t bytes) {
        char* p = w;
        w += (bytes + 63) & ~(size_t)63;
        return p;
    };
    _Float16* obs16   = (_Float16*)carve((size_t)MROWS * DDIM * 2);  // 16.8 MB
    _Float16* h1_16   = (_Float16*)carve((size_t)MROWS * H1D  * 2);  //  8.4 MB
    _Float16* full16  = (_Float16*)carve((size_t)MROWS * MSGD * 2);  //  4.2 MB
    _Float16* comp16  = (_Float16*)carve((size_t)MROWS * CDD  * 2);  //  2.1 MB
    _Float16* msgs16  = (_Float16*)carve((size_t)MROWS * MSGD * 2);  //  4.2 MB
    _Float16* tmp16   = (_Float16*)carve((size_t)MROWS * DDIM * 2);  // 16.8 MB
    float*    scores  = (float*)   carve((size_t)BDIM * NAG * NAG * 4); // 33.6 MB
    _Float16* agg16   = (_Float16*)carve((size_t)MROWS * MSGD * 2);  //  4.2 MB
    _Float16* r16     = tmp16;    // alias: tmp dead after scores
    // packed f16 weights (fragment-major)
    _Float16* W1p   = (_Float16*)carve((size_t)DDIM * H1D * 2);
    _Float16* W2p   = (_Float16*)carve((size_t)H1D * MSGD * 2);
    _Float16* Wcp   = (_Float16*)carve((size_t)MSGD * CDD * 2);
    _Float16* Wdp   = (_Float16*)carve((size_t)CDD * MSGD * 2);
    _Float16* Wbilp = (_Float16*)carve((size_t)DDIM * DDIM * 2);
    _Float16* Wr1p  = (_Float16*)carve((size_t)(DDIM + MSGD) * H2D * 2);
    _Float16* Wr2p  = (_Float16*)carve((size_t)H2D * DDIM * 2);

    const dim3 blk(256);
    const int NOSPLIT = 1 << 30;
    const int gx = MROWS / 128;   // 256 blocks along M

    // ---- one-time precision staging ----
    cvt_f32_to_f16<<<dim3((MROWS * DDIM) / 1024), blk, 0, stream>>>(obs, obs16);
    pack_b_frag<<<dim3((DDIM/32)*(H1D/16)),  dim3(32), 0, stream>>>(W1,   W1p,   H1D,  H1D/16);
    pack_b_frag<<<dim3((H1D/32)*(MSGD/16)),  dim3(32), 0, stream>>>(W2,   W2p,   MSGD, MSGD/16);
    pack_b_frag<<<dim3((MSGD/32)*(CDD/16)),  dim3(32), 0, stream>>>(Wc,   Wcp,   CDD,  CDD/16);
    pack_b_frag<<<dim3((CDD/32)*(MSGD/16)),  dim3(32), 0, stream>>>(Wd,   Wdp,   MSGD, MSGD/16);
    pack_b_frag<<<dim3((DDIM/32)*(DDIM/16)), dim3(32), 0, stream>>>(Wbil, Wbilp, DDIM, DDIM/16);
    pack_b_frag<<<dim3(((DDIM+MSGD)/32)*(H2D/16)), dim3(32), 0, stream>>>(Wr1, Wr1p, H2D, H2D/16);
    pack_b_frag<<<dim3((H2D/32)*(DDIM/16)),  dim3(32), 0, stream>>>(Wr2,  Wr2p,  DDIM, DDIM/16);

    // ---- msg_gen MLP + autoencoder ----
    wmma_gemm<4><<<dim3(gx, H1D/64), blk, 0, stream>>>(
        obs16, DDIM, obs16, DDIM, NOSPLIT, W1p, b1, h1_16, DDIM, H1D, FLAG_RELU | FLAG_F16O);
    wmma_gemm<4><<<dim3(gx, MSGD/64), blk, 0, stream>>>(
        h1_16, H1D, h1_16, H1D, NOSPLIT, W2p, b2, full16, H1D, MSGD, FLAG_F16O);
    wmma_gemm<2><<<dim3(gx, CDD/32), blk, 0, stream>>>(
        full16, MSGD, full16, MSGD, NOSPLIT, Wcp, bc, comp16, MSGD, CDD, FLAG_F16O);
    wmma_gemm<4><<<dim3(gx, MSGD/64), blk, 0, stream>>>(
        comp16, CDD, comp16, CDD, NOSPLIT, Wdp, bd, msgs16, CDD, MSGD, FLAG_F16O);

    // ---- bilinear relevance ----
    wmma_gemm<4><<<dim3(gx, DDIM/64), blk, 0, stream>>>(
        obs16, DDIM, obs16, DDIM, NOSPLIT, Wbilp, nullptr, tmp16, DDIM, DDIM, FLAG_F16O);
    wmma_scores<<<dim3(NAG/16/8, NAG/64, BDIM), blk, 0, stream>>>(
        tmp16, obs16, bbil, scores);

    // ---- top-8 + softmax + gather-aggregate ----
    topk_softmax_agg<<<dim3(MROWS / 8), blk, 0, stream>>>(scores, msgs16, agg16);

    // ---- receiver MLP on concat([obs, agg]) via split-K A operand ----
    wmma_gemm<4><<<dim3(gx, H2D/64), blk, 0, stream>>>(
        obs16, DDIM, agg16, MSGD, DDIM, Wr1p, br1, r16, DDIM + MSGD, H2D, FLAG_RELU | FLAG_F16O);
    wmma_gemm<4><<<dim3(gx, DDIM/64), blk, 0, stream>>>(
        r16, H2D, r16, H2D, NOSPLIT, Wr2p, br2, out, H2D, DDIM, 0);
}
